// HeterodyneToBaseband_55216099557554
// MI455X (gfx1250) — compile-verified, matrix operand order checked by reading, not run on hardware
//
#include <hip/hip_runtime.h>
#include <math.h>

// ---------------------------------------------------------------------------
// Heterodyne-to-baseband on MI455X (gfx1250, wave32, WMMA).
//
// IIR biquads truncated to FIRs (bandpass pole r~0.775 -> 256 taps, lowpass
// r~0.52 -> 128 taps; truncation error ~1e-25). Lowpass FIR and the 3:1 sinc
// resampler fuse into one 168-tap stride-3 FIR applied at output frames only.
// Each FIR tile is a Toeplitz GEMM on v_wmma_f32_16x16x32_f16:
//   D[16 out-times x 16 channels] += A[16x32 tap-Toeplitz] * B[32 x 16 chans]
// A-fragments are tile-invariant -> precomputed in WMMA register layout.
// Signals staged through LDS (zero-padded) -> hot loop is pure ds_load_b128
// + v_wmma, no divergent guards. Stage-B LDS fill uses async global->LDS
// copies (global_load_async_to_lds_b128 + s_wait_asynccnt).
// ---------------------------------------------------------------------------

typedef __attribute__((ext_vector_type(16))) _Float16 v16h;
typedef __attribute__((ext_vector_type(8)))  _Float16 v8h;
typedef __attribute__((ext_vector_type(8)))  float    v8f;
typedef __attribute__((__vector_size__(4 * sizeof(int)))) int v4i_raw;

#define K1        256                 // bandpass FIR taps
#define K2        128                 // lowpass FIR taps
#define RES_TAPS  41                  // sinc resampler taps (orig=3,new=1,width=19)
#define DLEN      (K2 + RES_TAPS - 1) // 168 fused lowpass+resample taps
#define NCA       9                   // stage-A K-chunks of 32
#define NCB       8                   // stage-B K-chunks of 32 (U0 = 3f0-160, 8-aligned)
#define SXA       424                 // LDS row stride (halves), stage A (212 dw -> 16 bank residues)
#define SXB       600                 // LDS row stride (halves), stage B (300 dw -> 16 bank residues)
#define WINA      400                 // valid window halves per row, stage A
#define WINB      592                 // valid window halves per row, stage B (74 x 16B)
#define WS_OFF_FA 0                   // stage-A A-fragments: 9*512 halves = 9216 B
#define WS_OFF_FB 9216                // stage-B A-fragments: 8*512 halves = 8192 B
#define WS_OFF_YM 20480               // f16 mixed signal: 32*T halves

#if defined(__HIP_DEVICE_COMPILE__) && __has_builtin(__builtin_amdgcn_global_load_async_to_lds_b128)
#define HAVE_ASYNC_LDS 1
#else
#define HAVE_ASYNC_LDS 0
#endif

// ---------------------------------------------------------------------------
// Setup: read scalar inputs on-device, design biquads, build truncated
// impulse responses, fuse lowpass+resampler, and emit both Toeplitz
// A-operand fragment tables directly in WMMA VGPR layout ([chunk][lane][16]).
// ---------------------------------------------------------------------------
__global__ void het_setup_kernel(const int* __restrict__ srp,
                                 const int* __restrict__ flp,
                                 const int* __restrict__ fhp,
                                 char* __restrict__ ws)
{
    if (blockIdx.x != 0 || threadIdx.x != 0) return;
    const double PI = 3.14159265358979323846;
    double sr = (double)srp[0];
    double fl = (double)flp[0], fh = (double)fhp[0];
    double center = 0.5 * (fl + fh);
    double bw = fh - fl;
    double Q = center / bw; if (Q < 0.5) Q = 0.5;

    // bandpass biquad -> truncated impulse response (recursion in f32, as ref)
    double w0 = 2.0 * PI * center / sr;
    double alpha = sin(w0) / (2.0 * Q);
    double a0 = 1.0 + alpha;
    float bb0 = (float)(alpha / a0);
    float bb2 = (float)(-alpha / a0);
    float ba1 = (float)(-2.0 * cos(w0) / a0);
    float ba2 = (float)((1.0 - alpha) / a0);
    float hbp[K1];
    {
        float x1 = 0.f, x2 = 0.f, y1 = 0.f, y2 = 0.f;
        for (int j = 0; j < K1; ++j) {
            float xn = (j == 0) ? 1.f : 0.f;
            float yn = bb0 * xn + bb2 * x2 - ba1 * y1 - ba2 * y2;
            x2 = x1; x1 = xn; y2 = y1; y1 = yn;
            hbp[j] = yn;
        }
    }

    // lowpass biquad (cutoff = 0.45 * 16000, Q = 0.707)
    double cut = 0.45 * 16000.0;
    double w0l = 2.0 * PI * cut / sr;
    double al  = sin(w0l) / (2.0 * 0.707);
    double a0l = 1.0 + al;
    float lb0 = (float)(((1.0 - cos(w0l)) * 0.5) / a0l);
    float lb1 = (float)((1.0 - cos(w0l)) / a0l);
    float lb2 = lb0;
    float la1 = (float)(-2.0 * cos(w0l) / a0l);
    float la2 = (float)((1.0 - al) / a0l);
    float hlp[K2];
    {
        float x1 = 0.f, x2 = 0.f, y1 = 0.f, y2 = 0.f;
        for (int j = 0; j < K2; ++j) {
            float xn = (j == 0) ? 1.f : 0.f;
            float yn = lb0 * xn + lb1 * x1 + lb2 * x2 - la1 * y1 - la2 * y2;
            x2 = x1; x1 = xn; y2 = y1; y1 = yn;
            hlp[j] = yn;
        }
    }

    // 3:1 polyphase sinc kernel (orig=3, new=1, width=19, rolloff 0.99)
    float kern[RES_TAPS];
    for (int i = 0; i < RES_TAPS; ++i) {
        double idx = (double)(i - 19) / 3.0;
        double t = idx * 0.99;
        if (t > 6.0) t = 6.0;
        if (t < -6.0) t = -6.0;
        double wdw = cos(t * PI / 12.0); wdw *= wdw;
        double tp = t * PI;
        double s = (tp == 0.0) ? 1.0 : (sin(tp) / tp);
        kern[i] = (float)(s * wdw * (0.99 / 3.0));
    }

    // fuse lowpass + resampler: out[f] = sum_d C[d] * ym[3f + 21 - d]
    float C[DLEN];
    for (int d = 0; d < DLEN; ++d) {
        float c = 0.f;
        for (int i = 0; i < RES_TAPS; ++i) {
            int j = d + i - 40;
            if (j >= 0 && j < K2) c += kern[i] * hlp[j];
        }
        C[d] = c;
    }

    // A-operand fragments, pre-swizzled to the ISA 16-bit A VGPR layout:
    // lane -> row m = lane&15, element i -> k = 8*(lane>>4) + (i&7) + 16*(i>=8)
    _Float16* FA = (_Float16*)(ws + WS_OFF_FA);
    for (int c = 0; c < NCA; ++c)
        for (int lane = 0; lane < 32; ++lane)
            for (int i = 0; i < 16; ++i) {
                int m = lane & 15, hi = lane >> 4;
                int k = (i & 7) + (hi << 3) + ((i >> 3) << 4);
                int j = m + 272 - 32 * c - k;       // stage-A Toeplitz index
                FA[(c * 32 + lane) * 16 + i] =
                    (j >= 0 && j < K1) ? (_Float16)hbp[j] : (_Float16)0.f;
            }
    _Float16* FB = (_Float16*)(ws + WS_OFF_FB);
    for (int c = 0; c < NCB; ++c)
        for (int lane = 0; lane < 32; ++lane)
            for (int i = 0; i < 16; ++i) {
                int m = lane & 15, hi = lane >> 4;
                int k = (i & 7) + (hi << 3) + ((i >> 3) << 4);
                int d = 3 * m + 181 - 32 * c - k;   // stride-3 Toeplitz index
                FB[(c * 32 + lane) * 16 + i] =
                    (d >= 0 && d < DLEN) ? (_Float16)C[d] : (_Float16)0.f;
            }
}

// ---------------------------------------------------------------------------
// Stage A: ym[ch,t] = cos(w*t) * (h_bp (*) x)[ch,t], stored f16.
// Block: 8 waves -> 8 consecutive 16-time tiles, one 16-channel group.
// ---------------------------------------------------------------------------
__global__ __launch_bounds__(256) void het_stageA(
    const float* __restrict__ x, char* __restrict__ ws,
    const int* __restrict__ srp, const int* __restrict__ flp,
    const int* __restrict__ fhp, int T)
{
    __shared__ __align__(16) _Float16 sFA[NCA * 512];
    __shared__ __align__(16) _Float16 sX[16 * SXA];
    const int tid = threadIdx.x;

    // stage tap fragments (dword copy)
    {
        const unsigned* gFA = (const unsigned*)(ws + WS_OFF_FA);
        unsigned* lFA = (unsigned*)sFA;
        for (int i = tid; i < NCA * 256; i += 256) lFA[i] = gFA[i];
    }

    // stage zero-padded f16 signal window: u in [span0-272, span0+127]
    const int grp = blockIdx.y;
    const int span0 = blockIdx.x * 128;
    const int W0A = span0 - 272;
    for (int idx = tid; idx < 16 * WINA; idx += 256) {
        int row = idx / WINA;
        int col = idx - row * WINA;
        int u = W0A + col;
        int uc = u < 0 ? 0 : (u >= T ? T - 1 : u);   // clamp, then select
        const float* xr = x + (size_t)(grp * 16 + row) * (size_t)T;
        float v = xr[uc];
        __builtin_prefetch(xr + uc + 512, 0, 1);     // global_prefetch_b8
        sX[row * SXA + col] = (u >= 0 && u < T) ? (_Float16)v : (_Float16)0.f;
    }
    __syncthreads();

    const int lane = tid & 31, wave = tid >> 5;
    const int hi = lane >> 4, mn = lane & 15;
    const int t0 = span0 + 16 * wave;
    if (t0 >= T) return;

    const float wh = (float)(2.0 * 3.14159265358979323846 *
                             (0.5 * (double)(flp[0] + fhp[0])) / (double)srp[0]);

    v8f acc = {};
#pragma unroll
    for (int c = 0; c < NCA; ++c) {
        v16h a = *(const v16h*)&sFA[(c * 32 + lane) * 16];       // 32B aligned
        int base = 16 * wave + 32 * c + (hi << 3);               // mult of 8
        v8h blo = *(const v8h*)&sX[mn * SXA + base];
        v8h bhi = *(const v8h*)&sX[mn * SXA + base + 16];
        v16h b = __builtin_shufflevector(blo, bhi,
                     0, 1, 2, 3, 4, 5, 6, 7, 8, 9, 10, 11, 12, 13, 14, 15);
        acc = __builtin_amdgcn_wmma_f32_16x16x32_f16(
                  false, a, false, b, (short)0, acc, false, false);
    }

    // modulate by cos(w*t); pack 8 consecutive times -> one 16B store
    _Float16* ym = (_Float16*)(ws + WS_OFF_YM);
    const int tb = t0 + (hi << 3);
    v8h o;
#pragma unroll
    for (int r = 0; r < 8; ++r)
        o[r] = (_Float16)(acc[r] * __cosf(wh * (float)(tb + r)));
    size_t off = (size_t)(grp * 16 + mn) * (size_t)T + tb;
    if (tb + 7 < T) {
        *(v8h*)&ym[off] = o;
    } else {
#pragma unroll
        for (int r = 0; r < 8; ++r)
            if (tb + r < T) ym[off + r] = o[r];
    }
}

// ---------------------------------------------------------------------------
// Stage B: fused lowpass + 3:1 resample.
// out[ch,f] = sum_d C[d] * ym[ch, 3f + 21 - d], window base 3*span0-160.
// ---------------------------------------------------------------------------
__global__ __launch_bounds__(256) void het_stageB(
    char* __restrict__ ws, float* __restrict__ out, int T, int F)
{
    __shared__ __align__(16) _Float16 sFB[NCB * 512];
    __shared__ __align__(16) _Float16 sY[16 * SXB];
    const int tid = threadIdx.x;

    {
        const unsigned* gFB = (const unsigned*)(ws + WS_OFF_FB);
        unsigned* lFB = (unsigned*)sFB;
        for (int i = tid; i < NCB * 256; i += 256) lFB[i] = gFB[i];
    }

    const int grp = blockIdx.y;
    const int span0 = blockIdx.x * 128;
    const int W0B = 3 * span0 - 160;                 // multiple of 8 -> 16B aligned
    const _Float16* __restrict__ ymg =
        (const _Float16*)(ws + WS_OFF_YM) + (size_t)(grp * 16) * (size_t)T;

    const bool interior = (W0B >= 0) && (W0B + WINB <= T);
    if (interior) {
#if HAVE_ASYNC_LDS
        // pure f16 copy -> async global->LDS (16B per lane per op)
        for (int it = 0; it < 5; ++it) {
            int idx = it * 256 + tid;
            if (idx < 16 * 74) {                     // 74 x 16B chunks per row
                int row = idx / 74;
                int col = (idx - row * 74) * 8;
                const void* gsrc = (const void*)(ymg + (size_t)row * T + W0B + col);
                __attribute__((address_space(1))) v4i_raw* gp =
                    (__attribute__((address_space(1))) v4i_raw*)(unsigned long long)(size_t)gsrc;
                __attribute__((address_space(3))) v4i_raw* lp =
                    (__attribute__((address_space(3))) v4i_raw*)(unsigned)(size_t)(void*)&sY[row * SXB + col];
                __builtin_amdgcn_global_load_async_to_lds_b128(gp, lp, 0, 0);
            }
        }
#if __has_builtin(__builtin_amdgcn_s_wait_asynccnt)
        __builtin_amdgcn_s_wait_asynccnt(0);
#else
        asm volatile("s_wait_asynccnt 0x0" ::: "memory");
#endif
#else
        for (int idx = tid; idx < 16 * 74; idx += 256) {
            int row = idx / 74;
            int col = (idx - row * 74) * 8;
            *(uint4*)&sY[row * SXB + col] =
                *(const uint4*)(ymg + (size_t)row * T + W0B + col);
        }
#endif
    } else {
        for (int idx = tid; idx < 16 * WINB; idx += 256) {
            int row = idx / WINB;
            int col = idx - row * WINB;
            int u = W0B + col;
            int uc = u < 0 ? 0 : (u >= T ? T - 1 : u);
            _Float16 v = ymg[(size_t)row * T + uc];
            sY[row * SXB + col] = (u >= 0 && u < T) ? v : (_Float16)0.f;
        }
    }
    __syncthreads();

    const int lane = tid & 31, wave = tid >> 5;
    const int hi = lane >> 4, mn = lane & 15;
    const int f0 = span0 + 16 * wave;
    if (f0 >= F) return;

    v8f acc = {};
#pragma unroll
    for (int c = 0; c < NCB; ++c) {
        v16h a = *(const v16h*)&sFB[(c * 32 + lane) * 16];       // 32B aligned
        int base = 48 * wave + 32 * c + (hi << 3);               // mult of 8
        v8h blo = *(const v8h*)&sY[mn * SXB + base];
        v8h bhi = *(const v8h*)&sY[mn * SXB + base + 16];
        v16h b = __builtin_shufflevector(blo, bhi,
                     0, 1, 2, 3, 4, 5, 6, 7, 8, 9, 10, 11, 12, 13, 14, 15);
        acc = __builtin_amdgcn_wmma_f32_16x16x32_f16(
                  false, a, false, b, (short)0, acc, false, false);
    }

    const int fb = f0 + (hi << 3);
    size_t off = (size_t)(grp * 16 + mn) * (size_t)F + fb;
    if (fb + 7 < F) {
        float4 o0; o0.x = acc[0]; o0.y = acc[1]; o0.z = acc[2]; o0.w = acc[3];
        float4 o1; o1.x = acc[4]; o1.y = acc[5]; o1.z = acc[6]; o1.w = acc[7];
        *(float4*)&out[off]     = o0;
        *(float4*)&out[off + 4] = o1;
    } else {
#pragma unroll
        for (int r = 0; r < 8; ++r)
            if (fb + r < F) out[off + r] = acc[r];
    }
}

// Second tuple element of the reference: the scalar sample rate (16000).
__global__ void het_tail(float* __restrict__ out, int idx)
{
    if (threadIdx.x == 0 && blockIdx.x == 0) out[idx] = 16000.0f;
}

// ---------------------------------------------------------------------------
extern "C" void kernel_launch(void* const* d_in, const int* in_sizes, int n_in,
                              void* d_out, int out_size, void* d_ws, size_t ws_size,
                              hipStream_t stream)
{
    const float* x  = (const float*)d_in[0];
    const int*   sr = (const int*)d_in[1];
    const int*   fl = (const int*)d_in[2];
    const int*   fh = (const int*)d_in[3];

    const int B = 32;
    const int T = in_sizes[0] / B;        // 480000
    const int F = out_size / B;           // 160000 (drops scalar tail if present)
    char*  ws  = (char*)d_ws;
    float* out = (float*)d_out;

    het_setup_kernel<<<1, 1, 0, stream>>>(sr, fl, fh, ws);

    const int spansA = (T + 127) / 128;
    dim3 gA(spansA, B / 16, 1);
    het_stageA<<<gA, 256, 0, stream>>>(x, ws, sr, fl, fh, T);

    const int spansB = (F + 127) / 128;
    dim3 gB(spansB, B / 16, 1);
    het_stageB<<<gB, 256, 0, stream>>>(ws, out, T, F);

    if (out_size == B * F + 1)
        het_tail<<<1, 1, 0, stream>>>(out, B * F);
}
